// QC_SlefAttention_25795573580582
// MI455X (gfx1250) — compile-verified, matrix operand order checked by reading, not run on hardware
//
#include <hip/hip_runtime.h>
#include <hip/hip_bf16.h>

// ---------------- problem constants ----------------
#define BQ      32
#define KSEQ    512
#define DMODEL  1024
#define NH      8
#define DHEAD   128
#define BKROWS  (BQ * KSEQ)      // 16384 rows
#define DFF     (4 * DMODEL)     // 4096
#define BN_EPS  1e-5f

// ---------------- vector types ----------------
typedef __attribute__((ext_vector_type(16))) __bf16        bf16x16;
typedef __attribute__((ext_vector_type(8)))  float         v8f;
typedef __attribute__((ext_vector_type(4)))  unsigned int  u32x4;
typedef __attribute__((ext_vector_type(4)))  float         f32x4;

union Frag16 { bf16x16 v; u32x4 q[2]; unsigned short u[16]; };

__device__ __forceinline__ unsigned short f2bf(float f) {
  union { float f; unsigned int u; } x; x.f = f;
  unsigned int r = x.u + 0x7fffu + ((x.u >> 16) & 1u);   // round-to-nearest-even
  return (unsigned short)(r >> 16);
}

// ---------------- WMMA GEMM ----------------
// Block tile 128x128, 8 waves (wave32), wave tile 32(M) x 64(N), K-step 32.
// A (MxK row-major fp32) staged into LDS as fp32 via GLOBAL_LOAD_ASYNC_TO_LDS
// (ASYNCcnt path, no VGPR round-trip); converted to bf16 at fragment build.
// B (KxN fp32, or NxK if TRANS_B) staged through VGPRs with fp32->bf16
// conversion (and LDS-side transpose for the NN case).
#define TBM 128
#define TBN 128
#define TKB 32
#define LDPA 36  // fp32 LDS pitch for A tile (32 + 4 pad, 16B aligned rows)
#define LDPB 40  // bf16 LDS pitch for B tile (32 + 8 pad, 16B aligned rows)

template<bool TRANS_B, bool HAS_BIAS, bool HAS_RES, bool RELU>
__global__ __launch_bounds__(256)
void gemm_wmma_bf16(const float* __restrict__ A, const float* __restrict__ Bm,
                    const float* __restrict__ bias, const float* __restrict__ Res,
                    float* __restrict__ C,
                    int M, int N, int Kd, int lda, int ldb, int ldc,
                    long long sAb, long long sAh, long long sBb, long long sBh,
                    long long sCb, long long sCh, int Hz)
{
  __shared__ float          Af[TBM * LDPA];   // fp32 A tile (18.4 KB)
  __shared__ unsigned short Bs[TBN * LDPB];   // bf16 B tile, [n][k] (10.2 KB)

  const int tid  = threadIdx.x;
  const int lane = tid & 31;
  const int wave = tid >> 5;
  const int half = lane >> 4;     // which 16-lane half
  const int r16  = lane & 15;

  const int z  = blockIdx.z;
  const int zb = z / Hz;
  const int zh = z - zb * Hz;
  A  += zb * sAb + zh * sAh;
  Bm += zb * sBb + zh * sBh;
  C  += zb * sCb + zh * sCh;
  const float* Rp = HAS_RES ? (Res + zb * sCb + zh * sCh) : nullptr;

  const int m0 = blockIdx.y * TBM;
  const int n0 = blockIdx.x * TBN;

  const int wm = (wave & 3) * 32;  // wave row offset inside block tile
  const int wn = (wave >> 2) * 64; // wave col offset inside block tile

  v8f acc[2][4];
  #pragma unroll
  for (int i = 0; i < 2; ++i)
    #pragma unroll
    for (int j = 0; j < 4; ++j)
      #pragma unroll
      for (int p = 0; p < 8; ++p) acc[i][j][p] = 0.0f;

  const int ksteps = Kd / TKB;
  for (int kt = 0; kt < ksteps; ++kt) {
    const int k0 = kt * TKB;

    // ---- stage A tile (TBM x 32 fp32) via async DMA: global -> LDS ----
    #pragma unroll
    for (int it = 0; it < 4; ++it) {
      int chunk = tid + it * 256;          // 1024 16B chunks
      int row   = chunk >> 3;              // 8 float4 per row
      int c4    = (chunk & 7) << 2;
      const float* gp = A + (long long)(m0 + row) * lda + k0 + c4;
      unsigned lds = (unsigned)(size_t)(&Af[row * LDPA + c4]);
      asm volatile("global_load_async_to_lds_b128 %0, %1, off"
                   :: "v"(lds), "v"((unsigned long long)(size_t)gp)
                   : "memory");
    }

    // ---- stage B tile into LDS as bf16 [n][k] (transpose for NN) ----
    if (!TRANS_B) {
      #pragma unroll
      for (int it = 0; it < 4; ++it) {
        int chunk = tid + it * 256;
        int krow  = chunk >> 5;            // 32 float4 per k-row (128 cols)
        int c4    = (chunk & 31) << 2;
        const float* gp = Bm + (long long)(k0 + krow) * ldb + n0 + c4;
        if (kt + 1 < ksteps)
          __builtin_prefetch(gp + (long long)TKB * ldb, 0, 3); // next k tile
        f32x4 v = *(const f32x4*)gp;
        #pragma unroll
        for (int j = 0; j < 4; ++j)
          Bs[(c4 + j) * LDPB + krow] = f2bf(v[j]);
      }
    } else {
      #pragma unroll
      for (int it = 0; it < 4; ++it) {
        int chunk = tid + it * 256;
        int nrow  = chunk >> 3;
        int c4    = (chunk & 7) << 2;
        const float* gp = Bm + (long long)(n0 + nrow) * ldb + k0 + c4;
        if (kt + 1 < ksteps)
          __builtin_prefetch(gp + TKB, 0, 3);                  // next k tile
        f32x4 v = *(const f32x4*)gp;
        unsigned int lo = (unsigned int)f2bf(v[0]) | ((unsigned int)f2bf(v[1]) << 16);
        unsigned int hi = (unsigned int)f2bf(v[2]) | ((unsigned int)f2bf(v[3]) << 16);
        unsigned short* d = &Bs[nrow * LDPB + c4];
        *(unsigned int*)(d)     = lo;
        *(unsigned int*)(d + 2) = hi;
      }
    }

    // async DMA completion (ASYNCcnt), then make LDS visible to all waves
    asm volatile("s_wait_asynccnt 0x0" ::: "memory");
    __syncthreads();

    // ---- fragments per ISA 7.12.2: elem e -> K = (e<8?0:16) + half*8 + e%8 ----
    Frag16 aF[2], bF[4];
    #pragma unroll
    for (int i = 0; i < 2; ++i) {
      const float* p = &Af[(wm + i * 16 + r16) * LDPA + half * 8];
      f32x4 a0 = *(const f32x4*)(p);       // K = half*8 .. +3
      f32x4 a1 = *(const f32x4*)(p + 4);   // K = half*8+4 .. +7
      f32x4 a2 = *(const f32x4*)(p + 16);  // K = 16+half*8 .. +3
      f32x4 a3 = *(const f32x4*)(p + 20);  // K = 16+half*8+4 .. +7
      #pragma unroll
      for (int e = 0; e < 4; ++e) {
        aF[i].u[e]      = f2bf(a0[e]);
        aF[i].u[4 + e]  = f2bf(a1[e]);
        aF[i].u[8 + e]  = f2bf(a2[e]);
        aF[i].u[12 + e] = f2bf(a3[e]);
      }
    }
    #pragma unroll
    for (int j = 0; j < 4; ++j) {
      const unsigned short* p = &Bs[(wn + j * 16 + r16) * LDPB + half * 8];
      bF[j].q[0] = *(const u32x4*)(p);
      bF[j].q[1] = *(const u32x4*)(p + 16);
    }

    #pragma unroll
    for (int i = 0; i < 2; ++i)
      #pragma unroll
      for (int j = 0; j < 4; ++j)
        acc[i][j] = __builtin_amdgcn_wmma_f32_16x16x32_bf16(
            false, aF[i].v, false, bF[j].v, (short)0, acc[i][j], false, false);
    __syncthreads();
  }

  // ---- epilogue: C layout VGPR p -> M = p + 8*half, N = lane%16 ----
  #pragma unroll
  for (int i = 0; i < 2; ++i) {
    #pragma unroll
    for (int j = 0; j < 4; ++j) {
      const int gcol = n0 + wn + j * 16 + r16;
      const float bv = HAS_BIAS ? bias[gcol] : 0.0f;
      #pragma unroll
      for (int p = 0; p < 8; ++p) {
        const int grow = m0 + wm + i * 16 + half * 8 + p;
        const long long idx = (long long)grow * ldc + gcol;
        float vv = acc[i][j][p] + bv;
        if (HAS_RES) vv += Rp[idx];
        if (RELU)    vv = fmaxf(vv, 0.0f);
        C[idx] = vv;
      }
    }
  }
}

// ---------------- masked scaled softmax over rows of length 512 ----------------
__global__ __launch_bounds__(256)
void softmax_mask_kernel(float* __restrict__ att, const unsigned char* __restrict__ mask)
{
  const int row = blockIdx.x;              // B*H*KSEQ rows
  const int b   = row >> 12;               // row / (NH*KSEQ)
  float* p = att + (long long)row * KSEQ;
  const unsigned char* mrow = mask + b * KSEQ;
  const int t = threadIdx.x;
  const float SCALE = 0.08838834764831845f;  // 1/sqrt(128)

  float v0 = mrow[t]       ? -1e9f : p[t]       * SCALE;
  float v1 = mrow[t + 256] ? -1e9f : p[t + 256] * SCALE;

  __shared__ float red[256];
  red[t] = fmaxf(v0, v1);
  __syncthreads();
  for (int s = 128; s > 0; s >>= 1) {
    if (t < s) red[t] = fmaxf(red[t], red[t + s]);
    __syncthreads();
  }
  const float mx = red[0];
  __syncthreads();
  const float e0 = __expf(v0 - mx), e1 = __expf(v1 - mx);
  red[t] = e0 + e1;
  __syncthreads();
  for (int s = 128; s > 0; s >>= 1) {
    if (t < s) red[t] += red[t + s];
    __syncthreads();
  }
  const float inv = 1.0f / red[0];
  p[t]       = e0 * inv;
  p[t + 256] = e1 * inv;
}

// ---------------- deterministic 2-pass BatchNorm over (BKROWS, D) ----------------
__global__ __launch_bounds__(256)
void bn_partial(const float* __restrict__ x, float* __restrict__ psum, float* __restrict__ psq)
{
  const int blk = blockIdx.x;              // 64 blocks * 256 rows
  const int t   = threadIdx.x;
  float s[4] = {0, 0, 0, 0}, q[4] = {0, 0, 0, 0};
  const int r0 = blk * 256;
  for (int r = 0; r < 256; ++r) {
    const float* xp = x + (long long)(r0 + r) * DMODEL;
    #pragma unroll
    for (int j = 0; j < 4; ++j) {
      float v = xp[t + j * 256];
      s[j] += v; q[j] += v * v;
    }
  }
  #pragma unroll
  for (int j = 0; j < 4; ++j) {
    psum[blk * DMODEL + t + j * 256] = s[j];
    psq [blk * DMODEL + t + j * 256] = q[j];
  }
}

__global__ __launch_bounds__(256)
void bn_finalize(const float* __restrict__ psum, const float* __restrict__ psq,
                 const float* __restrict__ gamma, const float* __restrict__ beta,
                 float* __restrict__ scale, float* __restrict__ shift)
{
  const int c = blockIdx.x * 256 + threadIdx.x;
  float s = 0.0f, q = 0.0f;
  for (int b = 0; b < 64; ++b) { s += psum[b * DMODEL + c]; q += psq[b * DMODEL + c]; }
  const float invN = 1.0f / (float)BKROWS;
  const float mean = s * invN;
  const float var  = q * invN - mean * mean;
  const float inv  = rsqrtf(var + BN_EPS);
  const float g    = gamma[c] * inv;
  scale[c] = g;
  shift[c] = beta[c] - mean * g;
}

__global__ __launch_bounds__(256)
void bn_apply(const float* __restrict__ x, const float* __restrict__ scale,
              const float* __restrict__ shift, float* __restrict__ y)
{
  long long i = (long long)blockIdx.x * 256 + threadIdx.x;
  const long long total = (long long)BKROWS * DMODEL;
  const long long step  = (long long)gridDim.x * 256;
  for (; i < total; i += step) {
    const int c = (int)(i & (DMODEL - 1));
    y[i] = x[i] * scale[c] + shift[c];
  }
}

// ---------------- launcher ----------------
extern "C" void kernel_launch(void* const* d_in, const int* in_sizes, int n_in,
                              void* d_out, int out_size, void* d_ws, size_t ws_size,
                              hipStream_t stream)
{
  const float*         embed = (const float*)d_in[0];
  const unsigned char* mask  = (const unsigned char*)d_in[1];
  const float* wq = (const float*)d_in[2];  const float* bq = (const float*)d_in[3];
  const float* wk = (const float*)d_in[4];  const float* bk = (const float*)d_in[5];
  const float* wv = (const float*)d_in[6];  const float* bv = (const float*)d_in[7];
  const float* wm = (const float*)d_in[8];  const float* bm = (const float*)d_in[9];
  const float* w1 = (const float*)d_in[10]; const float* b1 = (const float*)d_in[11];
  const float* w2 = (const float*)d_in[12]; const float* b2 = (const float*)d_in[13];
  const float* g1 = (const float*)d_in[14]; const float* be1 = (const float*)d_in[15];
  const float* g2 = (const float*)d_in[16]; const float* be2 = (const float*)d_in[17];

  float* out_x = (float*)d_out;
  float* att   = out_x + (long long)BKROWS * DMODEL;   // second tuple element

  const long long SLOT = (long long)BKROWS * DMODEL;   // 16,777,216 floats
  float* ws = (float*)d_ws;
  float* Q    = ws + 0 * SLOT;
  float* Kp   = ws + 1 * SLOT;
  float* V    = ws + 2 * SLOT;
  float* At   = ws + 3 * SLOT;  // atted
  float* Hh   = ws + 0 * SLOT;  // FFN hidden reuses slots 0..3 (Q/K/V/At dead)
  float* x0   = ws + 4 * SLOT;  // also x2 after BN1
  float* x1   = ws + 5 * SLOT;
  float* psum = ws + 6 * SLOT;
  float* psq  = psum + 64 * DMODEL;
  float* scl  = psq  + 64 * DMODEL;
  float* sft  = scl  + DMODEL;

  const dim3 blk(256);
  const dim3 gD (DMODEL / TBN, BKROWS / TBM, 1);            // M=16384,N=1024
  const dim3 gF (DFF    / TBN, BKROWS / TBM, 1);            // M=16384,N=4096
  const dim3 gS (KSEQ   / TBN, KSEQ   / TBM, BQ * NH);      // scores 512x512 x256
  const dim3 gV (DHEAD  / TBN, KSEQ   / TBM, BQ * NH);      // att@V 512x128 x256

  // Q/K/V projections
  gemm_wmma_bf16<false, true, false, false><<<gD, blk, 0, stream>>>(
      embed, wq, bq, nullptr, Q, BKROWS, DMODEL, DMODEL, DMODEL, DMODEL, DMODEL,
      0, 0, 0, 0, 0, 0, 1);
  gemm_wmma_bf16<false, true, false, false><<<gD, blk, 0, stream>>>(
      embed, wk, bk, nullptr, Kp, BKROWS, DMODEL, DMODEL, DMODEL, DMODEL, DMODEL,
      0, 0, 0, 0, 0, 0, 1);
  gemm_wmma_bf16<false, true, false, false><<<gD, blk, 0, stream>>>(
      embed, wv, bv, nullptr, V, BKROWS, DMODEL, DMODEL, DMODEL, DMODEL, DMODEL,
      0, 0, 0, 0, 0, 0, 1);

  // scores = Q @ K^T per (b,h)  [NT GEMM]
  gemm_wmma_bf16<true, false, false, false><<<gS, blk, 0, stream>>>(
      Q, Kp, nullptr, nullptr, att, KSEQ, KSEQ, DHEAD, DMODEL, DMODEL, KSEQ,
      (long long)KSEQ * DMODEL, DHEAD,
      (long long)KSEQ * DMODEL, DHEAD,
      (long long)NH * KSEQ * KSEQ, (long long)KSEQ * KSEQ, NH);

  // mask + scale + softmax (in place on att output)
  softmax_mask_kernel<<<dim3(BQ * NH * KSEQ), blk, 0, stream>>>(att, mask);

  // atted = att @ V per (b,h), written as [b,q,h,dh] (== [BK, D])
  gemm_wmma_bf16<false, false, false, false><<<gV, blk, 0, stream>>>(
      att, V, nullptr, nullptr, At, KSEQ, DHEAD, KSEQ, KSEQ, DMODEL, DMODEL,
      (long long)NH * KSEQ * KSEQ, (long long)KSEQ * KSEQ,
      (long long)KSEQ * DMODEL, DHEAD,
      (long long)KSEQ * DMODEL, DHEAD, NH);

  // x0 = atted @ wm + bm + embed
  gemm_wmma_bf16<false, true, true, false><<<gD, blk, 0, stream>>>(
      At, wm, bm, embed, x0, BKROWS, DMODEL, DMODEL, DMODEL, DMODEL, DMODEL,
      0, 0, 0, 0, 0, 0, 1);

  // BatchNorm 1 -> x1
  bn_partial <<<dim3(64), blk, 0, stream>>>(x0, psum, psq);
  bn_finalize<<<dim3(DMODEL / 256), blk, 0, stream>>>(psum, psq, g1, be1, scl, sft);
  bn_apply   <<<dim3(4096), blk, 0, stream>>>(x0, scl, sft, x1);

  // FFN: Hh = relu(x1 @ w1 + b1); x2 = Hh @ w2 + b2 + x1
  gemm_wmma_bf16<false, true, false, true><<<gF, blk, 0, stream>>>(
      x1, w1, b1, nullptr, Hh, BKROWS, DFF, DMODEL, DMODEL, DFF, DFF,
      0, 0, 0, 0, 0, 0, 1);
  float* x2 = x0;
  gemm_wmma_bf16<false, true, true, false><<<gD, blk, 0, stream>>>(
      Hh, w2, b2, x1, x2, BKROWS, DMODEL, DFF, DFF, DMODEL, DMODEL,
      0, 0, 0, 0, 0, 0, 1);

  // BatchNorm 2 -> out_x
  bn_partial <<<dim3(64), blk, 0, stream>>>(x2, psum, psq);
  bn_finalize<<<dim3(DMODEL / 256), blk, 0, stream>>>(psum, psq, g2, be2, scl, sft);
  bn_apply   <<<dim3(4096), blk, 0, stream>>>(x2, scl, sft, out_x);

  (void)in_sizes; (void)n_in; (void)out_size; (void)ws_size;
}
